// ScannedRNN_89593017794614
// MI455X (gfx1250) — matrix operand order repaired and checked by reading, not compiled, not run on hardware
//
#include <hip/hip_runtime.h>

// GRU scan: T=128, B=512, D=H=512.
// Per-step fused kernel: 6 bf16 WMMA GEMM chains (ir/iz/in from x, hr/hz/hn from h).
// All LDS tiles are stored in WMMA *fragment order* (lane-major, 48B lane stride,
// conflict-free for b128 reads) so every fragment is two ds_load_b128, no shuffles.
// Weights are pre-tiled once into fragment order in d_ws; per K-iter the step kernel
// pulls one contiguous 12KB block, via global_load_async_to_lds_b128 when available.

#define T_STEPS 128
#define BATCH   512
#define DIM     512
#define HID     512

// LDS subtile: 32 lanes x 16 bf16 (32B payload), lane stride 24 ushorts (48B)
// -> 48B*l: dword-bank = 12l mod 64, distinct for l=0..15 -> conflict-free b128.
#define LSTRIDE 24
#define SUBTILE (32 * LSTRIDE)   // 768 ushorts = 1536 B

typedef __attribute__((ext_vector_type(16))) __bf16        v16bf;
typedef __attribute__((ext_vector_type(8)))  float         v8f;
typedef __attribute__((ext_vector_type(8)))  unsigned int  v8u;
typedef __attribute__((ext_vector_type(4)))  int           v4i;

// async-to-LDS builtin takes (AS1 v4i*, AS3 v4i*, imm offset, imm cpol)
typedef __attribute__((address_space(1))) v4i* gv4i_p;
typedef __attribute__((address_space(3))) v4i* sv4i_p;

#if __has_builtin(__builtin_amdgcn_global_load_async_to_lds_b128)
#define HAVE_ASYNC_LDS 1
#endif

__device__ __forceinline__ unsigned short f2bf(float f) {
  unsigned int u = __float_as_uint(f);
  unsigned int r = u + 0x7FFFu + ((u >> 16) & 1u);
  return (unsigned short)(r >> 16);
}

// Fragment = 16 bf16 per lane, contiguous in LDS at p (16B aligned): 2x ds_load_b128.
__device__ __forceinline__ v16bf frag_from_lds(const unsigned short* p) {
  const uint4 a = *(const uint4*)p;
  const uint4 b = *(const uint4*)(p + 8);
  v8u u;
  u[0] = a.x; u[1] = a.y; u[2] = a.z; u[3] = a.w;
  u[4] = b.x; u[5] = b.y; u[6] = b.z; u[7] = b.w;
  return __builtin_bit_cast(v16bf, u);
}

// ---- one-time weight repack: fp32 [k][n] -> bf16 fragment-tiled ----
// Layout: block(bx=jpair 0..15, kt 0..15) of 6144 ushorts (12KB), each block:
//   [g(6)][jhalf(2)][lane(32)][e(16)]  with lane = n%16 + 16*(k16half), e = k%16.
__global__ void repack_weights_kernel(const float* __restrict__ Wi,
                                      const float* __restrict__ Whrz,
                                      const float* __restrict__ Whn,
                                      unsigned short* __restrict__ out) {
  const int total = DIM * 3 * HID;  // 786432 = blocks bx 0..7 only (cols 0..255)
  for (int i = blockIdx.x * blockDim.x + threadIdx.x; i < total;
       i += gridDim.x * blockDim.x) {
    const int blk = i / 6144, rem = i % 6144;
    const int bx = blk >> 4, kt = blk & 15;
    const int g  = rem / 1024;
    const int r2 = rem & 1023;
    const int jh = r2 >> 9;
    const int ln = (r2 >> 4) & 31;
    const int e  = r2 & 15;
    const int k   = kt * 32 + (ln >> 4) * 16 + e;
    const int col = bx * 32 + jh * 16 + (ln & 15);
    float v;
    if (g < 3)      v = Wi[(size_t)k * (3 * HID) + g * HID + col];
    else if (g < 5) v = Whrz[(size_t)k * (2 * HID) + (g - 3) * HID + col];
    else            v = Whn[(size_t)k * HID + col];
    out[i] = f2bf(v);
  }
}

__global__ __launch_bounds__(128) void gru_step_kernel(
    const float* __restrict__ x,            // [B,D]  ins + t*B*D
    const float* __restrict__ hprev,        // [B,H]  carry (ys[t-1] or h0)
    const unsigned char* __restrict__ rst,  // [B]
    const unsigned short* __restrict__ Wt,  // fragment-tiled bf16 weights
    const float* __restrict__ bi,           // [3H]
    const float* __restrict__ bhn,          // [H]
    float* __restrict__ yt,                 // [B,H]  ys + t*B*H
    float* __restrict__ hT,                 // [B,H]
    int last) {
  __shared__ __align__(16) unsigned short sX[2 * SUBTILE];   // rows 0-15 / 16-31
  __shared__ __align__(16) unsigned short sH[2 * SUBTILE];
  __shared__ __align__(16) unsigned short sW[12 * SUBTILE];  // [g][jhalf]

  const int tid  = threadIdx.x;
  const int lane = tid & 31;
  const int wave = tid >> 5;
  const int wm = wave >> 1, wn = wave & 1;   // 2x2 waves -> 32x32 output tile
  const int b0 = blockIdx.y * 32;
  const int j0 = blockIdx.x * 32;

  v8f acc[6];
#pragma unroll
  for (int g = 0; g < 6; ++g) acc[g] = v8f{};

  for (int kb = 0; kb < DIM; kb += 32) {
    // ---- weights: one contiguous 12KB block -> LDS (fragment order) ----
    const unsigned short* wsrc = Wt + (size_t)(blockIdx.x * 16 + (kb >> 5)) * 6144;
#pragma unroll
    for (int j = 0; j < 6; ++j) {
      const int c  = tid + 128 * j;              // 0..767 chunks of 16B
      const int g  = c >> 7;
      const int r2 = c & 127;
      const int jh = r2 >> 6;
      const int ln = (r2 & 63) >> 1;
      const int ch = r2 & 1;
      unsigned short* dst = &sW[(g * 2 + jh) * SUBTILE + ln * LSTRIDE + ch * 8];
#ifdef HAVE_ASYNC_LDS
      __builtin_amdgcn_global_load_async_to_lds_b128(
          (gv4i_p)(wsrc + (size_t)c * 8), (sv4i_p)dst, 0, 0);
#else
      *(uint4*)dst = *(const uint4*)(wsrc + (size_t)c * 8);
#endif
    }
    // ---- x / h tiles: fp32 -> bf16, written directly in fragment order ----
#pragma unroll
    for (int i = tid; i < 256; i += 128) {
      const int r  = i >> 3;                      // row 0..31
      const int c4 = (i & 7) * 4;                 // k 0..28 step 4
      const int mh = r >> 4, m16 = r & 15;
      const int hh = c4 >> 4;
      const int e0 = c4 & 15;
      const int off = (m16 + 16 * hh) * LSTRIDE + e0;
      const float4 xv = *(const float4*)(x + (size_t)(b0 + r) * DIM + kb + c4);
      unsigned short* dx = &sX[mh * SUBTILE + off];
      *(unsigned int*)(dx)     = (unsigned int)f2bf(xv.x) | ((unsigned int)f2bf(xv.y) << 16);
      *(unsigned int*)(dx + 2) = (unsigned int)f2bf(xv.z) | ((unsigned int)f2bf(xv.w) << 16);
      const float hm  = rst[b0 + r] ? 0.0f : 1.0f;
      const float4 hv = *(const float4*)(hprev + (size_t)(b0 + r) * HID + kb + c4);
      unsigned short* dh = &sH[mh * SUBTILE + off];
      *(unsigned int*)(dh)     = (unsigned int)f2bf(hv.x * hm) | ((unsigned int)f2bf(hv.y * hm) << 16);
      *(unsigned int*)(dh + 2) = (unsigned int)f2bf(hv.z * hm) | ((unsigned int)f2bf(hv.w * hm) << 16);
      if (kb + 32 < DIM) {                        // prefetch next K block
        __builtin_prefetch(x + (size_t)(b0 + r) * DIM + kb + 32 + c4, 0, 0);
        __builtin_prefetch(hprev + (size_t)(b0 + r) * HID + kb + 32 + c4, 0, 0);
      }
    }
#ifdef HAVE_ASYNC_LDS
#if __has_builtin(__builtin_amdgcn_s_wait_asynccnt)
    __builtin_amdgcn_s_wait_asynccnt(0);
#else
    asm volatile("s_wait_asynccnt 0" ::: "memory");
#endif
#endif
    __syncthreads();

    // ---- 6 WMMA chains; each fragment = 2x ds_load_b128, zero address math ----
    {
      const v16bf ax = frag_from_lds(&sX[wm * SUBTILE + lane * LSTRIDE]);
      const v16bf ah = frag_from_lds(&sH[wm * SUBTILE + lane * LSTRIDE]);
#pragma unroll
      for (int g = 0; g < 6; ++g) {
        const v16bf bw = frag_from_lds(&sW[(g * 2 + wn) * SUBTILE + lane * LSTRIDE]);
        acc[g] = __builtin_amdgcn_wmma_f32_16x16x32_bf16(
            false, (g < 3) ? ax : ah, false, bw, (short)0, acc[g], false, false);
      }
    }
    __syncthreads();
  }

  // ---- epilogue: gates + blend.  C/D layout: VGPR r -> M = r + 8*(lane>=16) ----
  const int hh  = lane >> 4;
  const int col = j0 + 16 * wn + (lane & 15);
  const float bir = bi[col];
  const float biz = bi[col + HID];
  const float bin = bi[col + 2 * HID];
  const float bn  = bhn[col];
#pragma unroll
  for (int r = 0; r < 8; ++r) {
    const int row = b0 + 16 * wm + r + 8 * hh;
    const float hp = rst[row] ? 0.0f : hprev[(size_t)row * HID + col];
    const float gr = 1.0f / (1.0f + __expf(-(acc[0][r] + bir + acc[3][r])));
    const float gz = 1.0f / (1.0f + __expf(-(acc[1][r] + biz + acc[4][r])));
    const float gn = tanhf(acc[2][r] + bin + gr * (acc[5][r] + bn));
    const float nh = (1.0f - gz) * gn + gz * hp;
    yt[(size_t)row * HID + col] = nh;
    if (last) hT[(size_t)row * HID + col] = nh;
  }
}

extern "C" void kernel_launch(void* const* d_in, const int* in_sizes, int n_in,
                              void* d_out, int out_size, void* d_ws, size_t ws_size,
                              hipStream_t stream) {
  const float* h0            = (const float*)d_in[0];
  const float* ins           = (const float*)d_in[1];
  const unsigned char* rsets = (const unsigned char*)d_in[2];  // jnp.bool_ = 1 byte
  const float* Wi            = (const float*)d_in[3];
  const float* bi            = (const float*)d_in[4];
  const float* Whrz          = (const float*)d_in[5];
  const float* Whn           = (const float*)d_in[6];
  const float* bhn           = (const float*)d_in[7];

  float* hT = (float*)d_out;                 // output order: (hT, ys)
  float* ys = hT + (size_t)BATCH * HID;

  unsigned short* Wt = (unsigned short*)d_ws;  // 3MB fragment-tiled bf16 weights

  repack_weights_kernel<<<1024, 256, 0, stream>>>(Wi, Whrz, Whn, Wt);

  const dim3 grid(HID / 32, BATCH / 32);       // 16x16 workgroups per step
  for (int t = 0; t < T_STEPS; ++t) {
    const float* xx    = ins + (size_t)t * BATCH * DIM;
    const float* hprev = (t == 0) ? h0 : (ys + (size_t)(t - 1) * BATCH * HID);
    gru_step_kernel<<<grid, 128, 0, stream>>>(
        xx, hprev, rsets + (size_t)t * BATCH, Wt, bi, bhn,
        ys + (size_t)t * BATCH * HID, hT, (t == T_STEPS - 1) ? 1 : 0);
  }
}
